// DeformableTransformer_59012850647440
// MI455X (gfx1250) — matrix operand order repaired and checked by reading, not compiled, not run on hardware
//
#include <hip/hip_runtime.h>
#include <hip/hip_bf16.h>

// ---------------- problem constants (from reference) ----------------
#define N_HEADS   8
#define N_POINTS  4
#define N_LEVELS  4
#define C_MODEL   512
#define FFN_DIM   1024
#define BATCH     2
#define S_TOTAL   8500                 // 80*80 + 40*40 + 20*20 + 10*10
#define M_TOTAL   (BATCH * S_TOTAL)    // 17000
#define LN_EPS    1e-5f

typedef __attribute__((ext_vector_type(8)))  float  v8f;
typedef __attribute__((ext_vector_type(16))) __bf16 v16bf;

__device__ __forceinline__ unsigned short f32_to_bf16_rne(float f) {
  union { float f; unsigned u; } v; v.f = f;
  unsigned u = v.u;
  u += 0x7FFFu + ((u >> 16) & 1u);   // round-to-nearest-even
  return (unsigned short)(u >> 16);
}

// ---------------- activation cast: src -> bf16, (src+pos) -> bf16 ----------------
__global__ void convert_act_kernel(const float* __restrict__ src,
                                   const float* __restrict__ pos,
                                   unsigned short* __restrict__ srcbf,
                                   unsigned short* __restrict__ qbf,
                                   long long n) {
  long long i = (long long)blockIdx.x * blockDim.x + threadIdx.x;
  if (i < n) {
    float s = src[i];
    srcbf[i] = f32_to_bf16_rne(s);
    qbf[i]   = f32_to_bf16_rne(s + pos[i]);
  }
}

// ---------------- plain f32 -> bf16 cast ----------------
__global__ void cast_bf16_kernel(const float* __restrict__ in,
                                 unsigned short* __restrict__ out,
                                 long long n) {
  long long i = (long long)blockIdx.x * blockDim.x + threadIdx.x;
  if (i < n) out[i] = f32_to_bf16_rne(in[i]);
}

// ---------------- weight transpose + cast: W[K,N] f32 -> Wt[N,K] bf16 ----------------
__global__ void wtrans_kernel(const float* __restrict__ W,
                              unsigned short* __restrict__ Wt,
                              int K, int N) {
  long long i = (long long)blockIdx.x * blockDim.x + threadIdx.x;
  if (i < (long long)K * N) {
    int k = (int)(i / N);
    int n = (int)(i - (long long)k * N);
    Wt[(size_t)n * K + k] = f32_to_bf16_rne(W[i]);
  }
}

// ---------------- bf16 WMMA GEMM: D[M,N] = A[M,K] * Bt[N,K]^T + bias ----------------
// Workgroup = 4 waves; each wave owns a 32(M) x 64(N) tile -> 128x64 per block.
// K-step 32: 2 A frags + 4 B frags -> 8 independent WMMAs per step.
// Software-pipelined with two statically-indexed register buffers (double
// buffering): loads for k-step n+1 are issued before the WMMAs of step n, so
// the backend's waitcnt insertion can leave the newer loads in flight instead
// of draining LOADcnt to 0 before every WMMA pair.
// REQUIRES K % 64 == 0 (true for all calls here: K = 512 or 1024).
//
// Out-of-range A rows are CLAMPED to row M-1 (branch-free inner loop):
// D row m depends only on A row m and the epilogue never stores m >= M.
//
// A fragment layout (ISA 7.12.2, 16-bit 16x32): lane(m=l&15, h=l>>4),
//   elements 0..7  -> K = k0 + 8h .. +7   (16B contiguous)
//   elements 8..15 -> K = k0+16+8h .. +7  (16B contiguous)
// B fragment (32x16): lane(n=l&15, h=l>>4), same K mapping as A (0..7 -> k0+8h, 8..15 -> k0+16+8h).
struct Frag { union { v16bf v; uint4 q[2]; }; };

template <bool RELU, bool OUT_BF16>
__global__ void __launch_bounds__(128)
gemm_bf16_wmma(const unsigned short* __restrict__ A,   // [M,K] bf16
               const unsigned short* __restrict__ Bt,  // [N,K] bf16 (pre-transposed)
               const float* __restrict__ bias,         // [N]
               void* __restrict__ D,                   // [M,N] f32 or bf16
               int M, int N, int K) {
  const int lane = threadIdx.x & 31;
  const int wave = threadIdx.x >> 5;
  const int half = lane >> 4;
  const int l15  = lane & 15;
  const int mBase = blockIdx.x * 128 + wave * 32;
  const int nBase = blockIdx.y * 64;

  v8f acc[8] = {};   // [mi*4 + t]

  // clamped A row pointers (branch-free inner loop)
  const int mR0 = mBase + l15;
  const int mR1 = mBase + 16 + l15;
  const unsigned short* aRow0 = A + (size_t)(mR0 < M ? mR0 : M - 1) * K;
  const unsigned short* aRow1 = A + (size_t)(mR1 < M ? mR1 : M - 1) * K;

  auto loadA = [&](Frag* dst, int k0) {
    dst[0].q[0] = *(const uint4*)(aRow0 + k0 + half * 8);
    dst[0].q[1] = *(const uint4*)(aRow0 + k0 + 16 + half * 8);
    dst[1].q[0] = *(const uint4*)(aRow1 + k0 + half * 8);
    dst[1].q[1] = *(const uint4*)(aRow1 + k0 + 16 + half * 8);
  };
  auto loadB = [&](Frag* dst, int k0) {
#pragma unroll
    for (int t = 0; t < 4; ++t) {
      const unsigned short* bp =
          Bt + (size_t)(nBase + t * 16 + l15) * K + k0 + half * 8;
      dst[t].q[0] = *(const uint4*)(bp);
      dst[t].q[1] = *(const uint4*)(bp + 16);
    }
  };
  auto mathStep = [&](const Frag* aF, const Frag* bF) {
#pragma unroll
    for (int t = 0; t < 4; ++t) {
      acc[t] = __builtin_amdgcn_wmma_f32_16x16x32_bf16(
          false, aF[0].v, false, bF[t].v, (short)0, acc[t], false, false);
      acc[4 + t] = __builtin_amdgcn_wmma_f32_16x16x32_bf16(
          false, aF[1].v, false, bF[t].v, (short)0, acc[4 + t], false, false);
    }
  };

  // software pipeline: buffers statically indexed (no scratch spill),
  // two 32-wide k-steps per trip.
  Frag a0[2], a1[2], b0[4], b1[4];
  loadA(a0, 0);
  loadB(b0, 0);
  int k0 = 0;
  for (; k0 + 64 < K; k0 += 64) {
    loadA(a1, k0 + 32);
    loadB(b1, k0 + 32);
    __builtin_prefetch(aRow0 + k0 + 96, 0, 3);   // pull A stream HBM->L2 ahead
    __builtin_prefetch(aRow1 + k0 + 96, 0, 3);
    mathStep(a0, b0);
    loadA(a0, k0 + 64);
    loadB(b0, k0 + 64);
    mathStep(a1, b1);
  }
  // epilogue: exactly two k-steps remain (K % 64 == 0)
  loadA(a1, k0 + 32);
  loadB(b1, k0 + 32);
  mathStep(a0, b0);
  mathStep(a1, b1);

  // Epilogue. D layout (ISA 7.12.2 32-bit C/D): vgpr r, lane l ->
  //   m = mBase + mi*16 + r + 8*(l>>4), n = nBase + 16*t + (l&15)
#pragma unroll
  for (int mi = 0; mi < 2; ++mi) {
#pragma unroll
    for (int t = 0; t < 4; ++t) {
      const int n = nBase + t * 16 + l15;
      const float bn = bias ? bias[n] : 0.0f;
#pragma unroll
      for (int r = 0; r < 8; ++r) {
        const int m = mBase + mi * 16 + r + 8 * half;
        if (m < M) {
          float val = acc[mi * 4 + t][r] + bn;
          if (RELU) val = fmaxf(val, 0.0f);
          if (OUT_BF16)
            ((unsigned short*)D)[(size_t)m * N + n] = f32_to_bf16_rne(val);
          else
            ((float*)D)[(size_t)m * N + n] = val;
        }
      }
    }
  }
}

// ---------------- softmax over 16 (L*P) per (token, head) ----------------
__global__ void softmax16_kernel(float* __restrict__ logits, int rows) {
  int i = blockIdx.x * blockDim.x + threadIdx.x;
  if (i >= rows) return;
  float* p = logits + (size_t)i * 16;
  float mx = p[0];
#pragma unroll
  for (int k = 1; k < 16; ++k) mx = fmaxf(mx, p[k]);
  float s = 0.f;
  float e[16];
#pragma unroll
  for (int k = 0; k < 16; ++k) { e[k] = __expf(p[k] - mx); s += e[k]; }
  float inv = 1.0f / s;
#pragma unroll
  for (int k = 0; k < 16; ++k) p[k] = e[k] * inv;
}

// ---------------- deformable bilinear sampling + head aggregation ----------------
// One wave32 per (token m, head). 32 lanes x float2 cover d=64.
// value (34.8 MB) is resident in the 192 MB L2, so gathers are L2-bound.
__global__ void __launch_bounds__(256)
ms_deform_sample(const float* __restrict__ value,  // [M, 512]
                 const float* __restrict__ off,    // [M, 256]
                 const float* __restrict__ attn,   // [M, 128] (softmaxed)
                 float* __restrict__ out) {        // [M, 512]
  const int wid  = blockIdx.x * (blockDim.x >> 5) + (threadIdx.x >> 5);
  const int lane = threadIdx.x & 31;
  if (wid >= M_TOTAL * N_HEADS) return;
  const int head = wid & (N_HEADS - 1);
  const int m    = wid >> 3;
  const int b    = m / S_TOTAL;
  const int s    = m - b * S_TOTAL;

  const int Hs[4]     = {80, 40, 20, 10};
  const int Ws[4]     = {80, 40, 20, 10};
  const int starts[4] = {0, 6400, 8000, 8400};

  // reference point from the query's own level (valid_ratios == 1)
  int ql = 3;
  if (s < 6400) ql = 0; else if (s < 8000) ql = 1; else if (s < 8400) ql = 2;
  const int local = s - starts[ql];
  const int qi = local / Ws[ql];
  const int qj = local - qi * Ws[ql];
  const float refx = (qj + 0.5f) / (float)Ws[ql];
  const float refy = (qi + 0.5f) / (float)Hs[ql];

  const float* offRow = off  + (size_t)m * 256 + head * 32;  // [L,P,2]
  const float* atRow  = attn + (size_t)m * 128 + head * 16;  // [L,P]

  float2 acc = make_float2(0.f, 0.f);

  for (int l = 0; l < N_LEVELS; ++l) {
    const int H = Hs[l], W = Ws[l];
    const float* vbase =
        value + ((size_t)b * S_TOTAL + starts[l]) * C_MODEL + head * 64;
#pragma unroll
    for (int p = 0; p < N_POINTS; ++p) {
      const int o = l * N_POINTS + p;
      const float lx = refx + offRow[o * 2 + 0] / (float)W;  // norm = (W, H)
      const float ly = refy + offRow[o * 2 + 1] / (float)H;
      const float a  = atRow[o];
      const float x  = lx * W - 0.5f;
      const float y  = ly * H - 0.5f;
      const float xf = floorf(x), yf = floorf(y);
      const float wx = x - xf, wy = y - yf;
      const int x0 = (int)xf, y0 = (int)yf;
#pragma unroll
      for (int c = 0; c < 4; ++c) {
        const int xi = x0 + (c & 1);
        const int yi = y0 + (c >> 1);
        const float wgt =
            ((c & 1) ? wx : 1.f - wx) * ((c >> 1) ? wy : 1.f - wy) * a;
        if (xi >= 0 && xi < W && yi >= 0 && yi < H) {
          const float2* vp =
              (const float2*)(vbase + ((size_t)yi * W + xi) * C_MODEL);
          const float2 v = vp[lane];
          acc.x += wgt * v.x;
          acc.y += wgt * v.y;
        }
      }
    }
  }
  float2* op = (float2*)(out + (size_t)m * C_MODEL + head * 64);
  op[lane] = acc;
}

// ---------------- fused residual + LayerNorm (optional f32 and bf16 outputs) ----------------
__global__ void __launch_bounds__(256)
ln_residual_kernel(const float* __restrict__ a, const float* __restrict__ r,
                   const float* __restrict__ g, const float* __restrict__ be,
                   float* __restrict__ outF, unsigned short* __restrict__ outB) {
  __shared__ float sbuf[256];
  const int row = blockIdx.x;
  const int t   = threadIdx.x;
  const size_t base = (size_t)row * C_MODEL;
  const float v0 = a[base + t]       + r[base + t];
  const float v1 = a[base + t + 256] + r[base + t + 256];
  sbuf[t] = v0 + v1;
  __syncthreads();
  for (int k = 128; k > 0; k >>= 1) {
    if (t < k) sbuf[t] += sbuf[t + k];
    __syncthreads();
  }
  const float mean = sbuf[0] * (1.0f / C_MODEL);
  __syncthreads();
  const float d0 = v0 - mean, d1 = v1 - mean;
  sbuf[t] = d0 * d0 + d1 * d1;
  __syncthreads();
  for (int k = 128; k > 0; k >>= 1) {
    if (t < k) sbuf[t] += sbuf[t + k];
    __syncthreads();
  }
  const float inv = rsqrtf(sbuf[0] * (1.0f / C_MODEL) + LN_EPS);
  const float y0 = d0 * inv * g[t]       + be[t];
  const float y1 = d1 * inv * g[t + 256] + be[t + 256];
  if (outF) { outF[base + t] = y0; outF[base + t + 256] = y1; }
  if (outB) {
    outB[base + t]       = f32_to_bf16_rne(y0);
    outB[base + t + 256] = f32_to_bf16_rne(y1);
  }
}

// ---------------- driver ----------------
extern "C" void kernel_launch(void* const* d_in, const int* in_sizes, int n_in,
                              void* d_out, int out_size, void* d_ws, size_t ws_size,
                              hipStream_t stream) {
  (void)in_sizes; (void)n_in; (void)out_size; (void)ws_size;

  const float* src    = (const float*)d_in[0];
  const float* pos    = (const float*)d_in[1];
  const float* w_off  = (const float*)d_in[2];
  const float* b_off  = (const float*)d_in[3];
  const float* w_attn = (const float*)d_in[4];
  const float* b_attn = (const float*)d_in[5];
  const float* w_val  = (const float*)d_in[6];
  const float* b_val  = (const float*)d_in[7];
  const float* w_outp = (const float*)d_in[8];
  const float* b_outp = (const float*)d_in[9];
  const float* ln1_g  = (const float*)d_in[10];
  const float* ln1_b  = (const float*)d_in[11];
  const float* w1     = (const float*)d_in[12];
  const float* b1     = (const float*)d_in[13];
  const float* w2     = (const float*)d_in[14];
  const float* b2     = (const float*)d_in[15];
  const float* ln2_g  = (const float*)d_in[16];
  const float* ln2_b  = (const float*)d_in[17];
  float* out = (float*)d_out;

  const int M = M_TOTAL, C = C_MODEL;

  // bump allocator over d_ws (~290 MB total)
  char* wp = (char*)d_ws;
  auto alloc = [&](size_t bytes) -> void* {
    void* p = (void*)wp;
    wp += (bytes + 255) & ~(size_t)255;
    return p;
  };
  unsigned short* srcbf  = (unsigned short*)alloc((size_t)M * C * 2);
  unsigned short* qbf    = (unsigned short*)alloc((size_t)M * C * 2);
  unsigned short* wvalt  = (unsigned short*)alloc((size_t)512 * 512 * 2);
  unsigned short* wofft  = (unsigned short*)alloc((size_t)256 * 512 * 2);
  unsigned short* wattnt = (unsigned short*)alloc((size_t)128 * 512 * 2);
  unsigned short* woutt  = (unsigned short*)alloc((size_t)512 * 512 * 2);
  unsigned short* w1t    = (unsigned short*)alloc((size_t)1024 * 512 * 2);
  unsigned short* w2t    = (unsigned short*)alloc((size_t)512 * 1024 * 2);
  float*          valuef = (float*)alloc((size_t)M * 512 * 4);
  float*          offf   = (float*)alloc((size_t)M * 256 * 4);
  float*          logits = (float*)alloc((size_t)M * 128 * 4);
  float*          aout   = (float*)alloc((size_t)M * 512 * 4);
  unsigned short* aobf   = (unsigned short*)alloc((size_t)M * 512 * 2);
  float*          projf  = (float*)alloc((size_t)M * 512 * 4);
  float*          xf     = (float*)alloc((size_t)M * 512 * 4);
  unsigned short* xbf    = (unsigned short*)alloc((size_t)M * 512 * 2);
  unsigned short* hbf    = (unsigned short*)alloc((size_t)M * 1024 * 2);
  float*          ffnf   = (float*)alloc((size_t)M * 512 * 4);

  const long long MC = (long long)M * C;
  const int cvtBlocks = (int)((MC + 255) / 256);

  // 1) activation casts
  convert_act_kernel<<<cvtBlocks, 256, 0, stream>>>(src, pos, srcbf, qbf, MC);

  // 2) weight transposes + casts (Wt is [N,K] bf16 so WMMA B frags are contiguous)
  wtrans_kernel<<<(512 * 512 + 255) / 256, 256, 0, stream>>>(w_val, wvalt, 512, 512);
  wtrans_kernel<<<(512 * 256 + 255) / 256, 256, 0, stream>>>(w_off, wofft, 512, 256);
  wtrans_kernel<<<(512 * 128 + 255) / 256, 256, 0, stream>>>(w_attn, wattnt, 512, 128);
  wtrans_kernel<<<(512 * 512 + 255) / 256, 256, 0, stream>>>(w_outp, woutt, 512, 512);
  wtrans_kernel<<<(512 * 1024 + 255) / 256, 256, 0, stream>>>(w1, w1t, 512, 1024);
  wtrans_kernel<<<(1024 * 512 + 255) / 256, 256, 0, stream>>>(w2, w2t, 1024, 512);

  const int gm = (M + 127) / 128;

  // 3) value = src @ w_val + b_val
  gemm_bf16_wmma<false, false><<<dim3(gm, 512 / 64), 128, 0, stream>>>(
      srcbf, wvalt, b_val, valuef, M, 512, 512);
  // 4) offsets = q @ w_off + b_off
  gemm_bf16_wmma<false, false><<<dim3(gm, 256 / 64), 128, 0, stream>>>(
      qbf, wofft, b_off, offf, M, 256, 512);
  // 5) attn logits = q @ w_attn + b_attn
  gemm_bf16_wmma<false, false><<<dim3(gm, 128 / 64), 128, 0, stream>>>(
      qbf, wattnt, b_attn, logits, M, 128, 512);

  // 6) softmax over L*P per (token, head)
  softmax16_kernel<<<(M * N_HEADS + 255) / 256, 256, 0, stream>>>(logits, M * N_HEADS);

  // 7) bilinear sampling + aggregation -> [M,512]
  ms_deform_sample<<<(M * N_HEADS + 7) / 8, 256, 0, stream>>>(valuef, offf, logits, aout);

  // 8) out projection
  cast_bf16_kernel<<<cvtBlocks, 256, 0, stream>>>(aout, aobf, MC);
  gemm_bf16_wmma<false, false><<<dim3(gm, 512 / 64), 128, 0, stream>>>(
      aobf, woutt, b_outp, projf, M, 512, 512);

  // 9) x = LN1(src + proj), emit f32 (residual) + bf16 (FFN1 A)
  ln_residual_kernel<<<M, 256, 0, stream>>>(src, projf, ln1_g, ln1_b, xf, xbf);

  // 10) FFN1 with fused ReLU, emit bf16 directly for FFN2's A operand
  gemm_bf16_wmma<true, true><<<dim3(gm, 1024 / 64), 128, 0, stream>>>(
      xbf, w1t, b1, hbf, M, 1024, 512);
  // 11) FFN2
  gemm_bf16_wmma<false, false><<<dim3(gm, 512 / 64), 128, 0, stream>>>(
      hbf, w2t, b2, ffnf, M, 512, 1024);

  // 12) out = LN2(x + ffn)
  ln_residual_kernel<<<M, 256, 0, stream>>>(xf, ffnf, ln2_g, ln2_b, out,
                                            (unsigned short*)nullptr);
}
